// Q_network_4621384810937
// MI455X (gfx1250) — matrix operand order repaired and checked by reading, not compiled, not run on hardware
//
#include <hip/hip_runtime.h>
#include <hip/hip_bf16.h>
#include <math.h>

typedef __attribute__((ext_vector_type(16))) __bf16 v16bf;
typedef __attribute__((ext_vector_type(8)))  float  v8f;
typedef __attribute__((ext_vector_type(4)))  int    v4i;

// Problem constants (from reference setup_inputs)
#define BB 512
#define NN 32
#define DD 128
#define AA 16
#define HH 256
#define CC 128
#define TT (BB*NN)          // 16384 tokens
#define KSAE 160            // 144 (D+A) padded to multiple of 32

// GEMM block tile
#define GT_M 128            // 8 waves x 16 rows
#define GT_N 64             // 4 x 16-col subtiles per wave

// ---------------------------------------------------------------------------
// CDNA5 async global->LDS copy (GLOBAL_LOAD_ASYNC_TO_LDS_B128, ASYNCcnt).
// Builtin signature (from hipcc diagnostic): (v4i* AS1, v4i* AS3, imm, imm).
// ---------------------------------------------------------------------------
#if __has_builtin(__builtin_amdgcn_global_load_async_to_lds_b128)
#define QN_HAS_ASYNC_LDS 1
#endif

typedef __attribute__((address_space(1))) v4i* qn_gp_t;
typedef __attribute__((address_space(3))) v4i* qn_lp_t;

__device__ __forceinline__ void qn_copy16(const void* g, void* l) {
#ifdef QN_HAS_ASYNC_LDS
  __builtin_amdgcn_global_load_async_to_lds_b128((qn_gp_t)g, (qn_lp_t)l, 0, 0);
#else
  *(uint4*)l = *(const uint4*)g;
#endif
}

__device__ __forceinline__ void qn_async_join() {
#ifdef QN_HAS_ASYNC_LDS
#if __has_builtin(__builtin_amdgcn_s_wait_asynccnt)
  __builtin_amdgcn_s_wait_asynccnt(0);
#else
  asm volatile("s_wait_asynccnt 0x0" ::: "memory");
#endif
#endif
  __syncthreads();
}

// ---------------------------------------------------------------------------
// Build bf16 activations: states_bf [T,128], obsact_bf [T,160] (=[states,actions,0pad])
// ---------------------------------------------------------------------------
__global__ void qn_build_acts(const float* __restrict__ states,
                              const float* __restrict__ actions,
                              __bf16* __restrict__ states_bf,
                              __bf16* __restrict__ obsact_bf) {
  long i = (long)blockIdx.x * 256 + threadIdx.x;
  if (i < (long)TT * DD) states_bf[i] = (__bf16)states[i];
  if (i < (long)TT * KSAE) {
    long t = i / KSAE; int c = (int)(i % KSAE);
    float v = (c < DD) ? states[t * DD + c]
            : (c < DD + AA) ? actions[t * AA + (c - DD)] : 0.0f;
    obsact_bf[i] = (__bf16)v;
  }
}

// ---------------------------------------------------------------------------
// Convert + transpose weight: W [K,N] f32 -> Wt [N,Kp] bf16 (zero padded K->Kp)
// ---------------------------------------------------------------------------
__global__ void qn_wconv(const float* __restrict__ W, __bf16* __restrict__ Wt,
                         int K, int N, int Kp) {
  int i = blockIdx.x * 256 + threadIdx.x;
  if (i >= N * Kp) return;
  int n = i / Kp, k = i % Kp;
  Wt[i] = (k < K) ? (__bf16)W[(size_t)k * N + n] : (__bf16)0.0f;
}

// ---------------------------------------------------------------------------
// Block-tiled WMMA GEMM: C[M,N] = act(A[M,K]bf16 @ Bt[N,K]^T bf16 + bias)
// 256 threads = 8 waves; block tile 128x64; B panel (64 x K bf16, <=32KB)
// async-staged to LDS and shared by all waves; each wave keeps one A fragment
// and feeds 4 WMMAs with it (4 N-subtiles). K multiple of 32, K<=256.
// ---------------------------------------------------------------------------
__global__ __launch_bounds__(256)
void qn_gemm(const __bf16* __restrict__ A, const __bf16* __restrict__ Bt,
             const float* __restrict__ bias, void* __restrict__ Cp,
             int M, int N, int K, int act, int out_bf16) {
  __shared__ __bf16 Blds[GT_N * 256];            // 32KB max (64 rows x K)

  const int tid  = threadIdx.x;
  const int lane = tid & 31;
  const int wave = tid >> 5;
  const int l15  = lane & 15;
  const int h    = lane >> 4;
  const int n0   = blockIdx.x * GT_N;
  const int m0   = blockIdx.y * GT_M + wave * 16;

  // stage B panel (rows n0..n0+63 of Bt are contiguous: 64*K elements)
  {
    const __bf16* Bp = Bt + (size_t)n0 * K;
    for (int i = tid * 8; i < GT_N * K; i += 256 * 8)
      qn_copy16(Bp + i, Blds + i);
  }
  qn_async_join();

  v8f c0 = {}, c1 = {}, c2 = {}, c3 = {};
  const int m = m0 + l15;
  for (int k0 = 0; k0 < K; k0 += 32) {
    v16bf a;
#pragma unroll
    for (int v = 0; v < 8; ++v) {                // A: 16x32, ISA 16-bit layout
      int kk = ((v < 4) ? 0 : 16) + h * 8 + (v & 3) * 2;
      const __bf16* p = A + (size_t)m * K + k0 + kk;
      a[2 * v] = p[0]; a[2 * v + 1] = p[1];
    }
    v16bf b0, b1, b2, b3;
#pragma unroll
    for (int v = 0; v < 8; ++v) {                // B frags from LDS panel
      int kk = ((v < 4) ? 0 : 16) + h * 8 + (v & 3) * 2;
      const __bf16* p0 = Blds + (0 * 16 + l15) * K + k0 + kk;
      const __bf16* p1 = Blds + (1 * 16 + l15) * K + k0 + kk;
      const __bf16* p2 = Blds + (2 * 16 + l15) * K + k0 + kk;
      const __bf16* p3 = Blds + (3 * 16 + l15) * K + k0 + kk;
      b0[2 * v] = p0[0]; b0[2 * v + 1] = p0[1];
      b1[2 * v] = p1[0]; b1[2 * v + 1] = p1[1];
      b2[2 * v] = p2[0]; b2[2 * v + 1] = p2[1];
      b3[2 * v] = p3[0]; b3[2 * v + 1] = p3[1];
    }
    c0 = __builtin_amdgcn_wmma_f32_16x16x32_bf16(false, a, false, b0, (short)0, c0, false, false);
    c1 = __builtin_amdgcn_wmma_f32_16x16x32_bf16(false, a, false, b1, (short)0, c1, false, false);
    c2 = __builtin_amdgcn_wmma_f32_16x16x32_bf16(false, a, false, b2, (short)0, c2, false, false);
    c3 = __builtin_amdgcn_wmma_f32_16x16x32_bf16(false, a, false, b3, (short)0, c3, false, false);
  }

#pragma unroll
  for (int s = 0; s < 4; ++s) {
    const v8f c = (s == 0) ? c0 : (s == 1) ? c1 : (s == 2) ? c2 : c3;
    const int n = n0 + s * 16 + l15;
    const float bv = bias ? bias[n] : 0.0f;
#pragma unroll
    for (int v = 0; v < 8; ++v) {
      int mm = m0 + h * 8 + v;
      float val = c[v] + bv;
      if (act) val = tanhf(val);
      if (out_bf16) ((__bf16*)Cp)[(size_t)mm * N + n] = (__bf16)val;
      else          ((float*)Cp)[(size_t)mm * N + n] = val;
    }
  }
}

// ---------------------------------------------------------------------------
// Per-batch attention + MLP head.  One block per batch (256 threads = 8 waves).
// ---------------------------------------------------------------------------
__global__ __launch_bounds__(256)
void qn_attn(const __bf16* __restrict__ queryv,  // [T,256]
             const __bf16* __restrict__ keyv,    // [T,256]
             const float*  __restrict__ g,       // [T,256]
             const float*  __restrict__ curr,    // [T,128]
             const __bf16* __restrict__ f1_wT,   // [128,384]
             const float*  __restrict__ f1_b,    // [128]
             const __bf16* __restrict__ f2_wT,   // [16,128]
             const float*  __restrict__ f2_b,    // [16]
             const float*  __restrict__ actions, // [B,32,16]
             const float*  __restrict__ policies,// [B,32,16]
             float* __restrict__ out_value,      // [B,32,32]
             float* __restrict__ out_qval,       // [B,32]
             float* __restrict__ out_wfull) {    // [B,32,32]
  __shared__ __bf16 qkbuf[2 * NN * HH];  // 32KB: phase1 Q|K, phase2 h|hh
  __shared__ float  S[NN * NN];          // scores -> softmax weights
  __shared__ float  Qm[NN * AA];         // Q head output

  const int b    = blockIdx.x;
  const int tid  = threadIdx.x;
  const int lane = tid & 31;
  const int wave = tid >> 5;
  const int l15  = lane & 15;
  const int h    = lane >> 4;
  const long tok0 = (long)b * NN;

  __bf16* Qs = qkbuf;
  __bf16* Ks = qkbuf + NN * HH;

  // async-stage Q/K into LDS (each 32x256 bf16 = contiguous 16KB)
  for (int i = tid * 8; i < NN * HH; i += 256 * 8) {
    qn_copy16(queryv + tok0 * HH + i, Qs + i);
    qn_copy16(keyv   + tok0 * HH + i, Ks + i);
  }
  qn_async_join();

  // ---- scores: S = (Q @ K^T)/16, 2x2 WMMA tiles on waves 0..3 --------------
  if (wave < 4) {
    const int tm = (wave >> 1) * 16, tn = (wave & 1) * 16;
    v8f c = {};
    for (int k0 = 0; k0 < HH; k0 += 32) {
      v16bf a, bb;
      int m = tm + l15;
#pragma unroll
      for (int v = 0; v < 8; ++v) {
        int kk = ((v < 4) ? 0 : 16) + h * 8 + (v & 3) * 2;
        a[2 * v]     = Qs[m * HH + k0 + kk];
        a[2 * v + 1] = Qs[m * HH + k0 + kk + 1];
      }
      int n = tn + l15;
#pragma unroll
      for (int v = 0; v < 8; ++v) {
        int kk = ((v < 4) ? 0 : 16) + h * 8 + (v & 3) * 2;
        bb[2 * v]     = Ks[n * HH + k0 + kk];
        bb[2 * v + 1] = Ks[n * HH + k0 + kk + 1];
      }
      c = __builtin_amdgcn_wmma_f32_16x16x32_bf16(false, a, false, bb,
                                                  (short)0, c, false, false);
    }
    int n = tn + l15;
#pragma unroll
    for (int v = 0; v < 8; ++v) S[(tm + h * 8 + v) * NN + n] = c[v] * 0.0625f;
  }
  __syncthreads();

  // ---- masked softmax rows (exclude self), emit weights_full ---------------
  if (tid < NN) {
    const int i = tid;
    float mx = -3.4e38f;
    for (int j = 0; j < NN; ++j) if (j != i) mx = fmaxf(mx, S[i * NN + j]);
    float w[NN]; float sum = 0.0f;
    for (int j = 0; j < NN; ++j) {
      float e = (j == i) ? 0.0f : __expf(S[i * NN + j] - mx);
      w[j] = e; sum += e;
    }
    float inv = 1.0f / sum;
    for (int j = 0; j < NN; ++j) {
      float wv = (j == i) ? 0.0f : w[j] * inv;
      S[i * NN + j] = wv;
      out_wfull[((long)b * NN + i) * NN + j] = (j == i) ? 1.0f : wv;
    }
  }
  __syncthreads();

  // ---- node_feat = W @ g ; build h = [curr | node_feat] as bf16 ------------
  __bf16* hbuf = qkbuf;                     // 32 x 384 bf16 (24KB, reuses Q/K)
  {
    const int d = tid;                      // node_feat column
    float gv[NN];
    for (int j = 0; j < NN; ++j) gv[j] = g[(tok0 + j) * HH + d];
    for (int i = 0; i < NN; ++i) {
      float acc = 0.0f;
#pragma unroll
      for (int j = 0; j < NN; ++j) acc += S[i * NN + j] * gv[j];
      hbuf[i * (CC + HH) + CC + d] = (__bf16)acc;
    }
    if (tid < CC)
      for (int i = 0; i < NN; ++i)
        hbuf[i * (CC + HH) + tid] = (__bf16)curr[(tok0 + i) * CC + tid];
  }
  __syncthreads();

  // ---- f1: hh = tanh(h[32x384] @ f1_w[384x128] + b), WMMA 2x8 tiles --------
  __bf16* hh = qkbuf + NN * (CC + HH);      // 32 x 128 bf16
  const int KH = CC + HH;                   // 384
  for (int t = wave; t < 16; t += 8) {
    const int tm = (t >> 3) * 16, tn = (t & 7) * 16;
    v8f c = {};
    for (int k0 = 0; k0 < KH; k0 += 32) {
      v16bf a, bb;
      int m = tm + l15;
#pragma unroll
      for (int v = 0; v < 8; ++v) {
        int kk = ((v < 4) ? 0 : 16) + h * 8 + (v & 3) * 2;
        a[2 * v]     = hbuf[m * KH + k0 + kk];
        a[2 * v + 1] = hbuf[m * KH + k0 + kk + 1];
      }
      int n = tn + l15;
#pragma unroll
      for (int v = 0; v < 8; ++v) {
        int kk = ((v < 4) ? 0 : 16) + h * 8 + (v & 3) * 2;
        const __bf16* p = f1_wT + (size_t)n * KH + k0 + kk;
        bb[2 * v] = p[0]; bb[2 * v + 1] = p[1];
      }
      c = __builtin_amdgcn_wmma_f32_16x16x32_bf16(false, a, false, bb,
                                                  (short)0, c, false, false);
    }
    int n = tn + l15;
#pragma unroll
    for (int v = 0; v < 8; ++v)
      hh[(tm + h * 8 + v) * CC + n] = (__bf16)tanhf(c[v] + f1_b[n]);
  }
  __syncthreads();

  // ---- f2: Qm[32x16] = hh @ f2_w + b ---------------------------------------
  for (int o = tid; o < NN * AA; o += 256) {
    int i = o >> 4, aidx = o & 15;
    float acc = f2_b[aidx];
    for (int k = 0; k < CC; ++k)
      acc += (float)hh[i * CC + k] * (float)f2_wT[aidx * CC + k];
    Qm[o] = acc;
  }
  __syncthreads();

  // ---- Q_value = sum(actions * Q) ------------------------------------------
  if (tid < NN) {
    float acc = 0.0f;
    for (int a = 0; a < AA; ++a)
      acc += actions[(tok0 + tid) * AA + a] * Qm[tid * AA + a];
    out_qval[(long)b * NN + tid] = acc;
  }
  // ---- Value[i][m] = sum_a Q[i][a] * policies[m][a] ------------------------
  for (int o = tid; o < NN * NN; o += 256) {
    int i = o >> 5, m = o & 31;
    float acc = 0.0f;
#pragma unroll
    for (int a = 0; a < AA; ++a)
      acc += Qm[i * AA + a] * policies[(tok0 + m) * AA + a];
    out_value[((long)b * NN + i) * NN + m] = acc;
  }
}

// ---------------------------------------------------------------------------
extern "C" void kernel_launch(void* const* d_in, const int* in_sizes, int n_in,
                              void* d_out, int out_size, void* d_ws, size_t ws_size,
                              hipStream_t stream) {
  const float* states   = (const float*)d_in[0];
  const float* policies = (const float*)d_in[1];
  const float* actions  = (const float*)d_in[2];
  const float* se_w  = (const float*)d_in[3];  const float* se_b  = (const float*)d_in[4];
  const float* key_w = (const float*)d_in[5];  const float* key_b = (const float*)d_in[6];
  const float* q_w   = (const float*)d_in[7];  const float* q_b   = (const float*)d_in[8];
  const float* sae_w = (const float*)d_in[9];  const float* sae_b = (const float*)d_in[10];
  const float* av_w  = (const float*)d_in[11]; const float* av_b  = (const float*)d_in[12];
  const float* ca_w  = (const float*)d_in[13]; const float* ca_b  = (const float*)d_in[14];
  const float* f1_w  = (const float*)d_in[15]; const float* f1_b  = (const float*)d_in[16];
  const float* f2_w  = (const float*)d_in[17]; const float* f2_b  = (const float*)d_in[18];

  char* ws = (char*)d_ws;
  size_t off = 0;
  auto alloc = [&](size_t bytes) -> void* {
    void* p = ws + off;
    off += (bytes + 255) & ~(size_t)255;
    return p;
  };
  __bf16* states_bf = (__bf16*)alloc((size_t)TT * DD * 2);
  __bf16* obsact_bf = (__bf16*)alloc((size_t)TT * KSAE * 2);
  __bf16* s_embed   = (__bf16*)alloc((size_t)TT * HH * 2);
  __bf16* t1        = (__bf16*)alloc((size_t)TT * HH * 2);
  __bf16* keyv      = (__bf16*)alloc((size_t)TT * HH * 2);
  __bf16* queryv    = (__bf16*)alloc((size_t)TT * HH * 2);
  float*  gbuf      = (float*) alloc((size_t)TT * HH * 4);
  float*  currb     = (float*) alloc((size_t)TT * CC * 4);
  __bf16* se_wT  = (__bf16*)alloc((size_t)HH * DD * 2);
  __bf16* key_wT = (__bf16*)alloc((size_t)HH * HH * 2);
  __bf16* q_wT   = (__bf16*)alloc((size_t)HH * HH * 2);
  __bf16* sae_wT = (__bf16*)alloc((size_t)HH * KSAE * 2);
  __bf16* av_wT  = (__bf16*)alloc((size_t)HH * HH * 2);
  __bf16* ca_wT  = (__bf16*)alloc((size_t)CC * DD * 2);
  __bf16* f1_wT  = (__bf16*)alloc((size_t)CC * (CC + HH) * 2);
  __bf16* f2_wT  = (__bf16*)alloc((size_t)AA * CC * 2);
  (void)ws_size; (void)in_sizes; (void)n_in; (void)out_size;

  // 1) activations -> bf16
  {
    long total = (long)TT * KSAE;
    qn_build_acts<<<dim3((unsigned)((total + 255) / 256)), dim3(256), 0, stream>>>(
        states, actions, states_bf, obsact_bf);
  }
  // 2) weights -> bf16 transposed (padded K where needed)
  auto wconv = [&](const float* W, __bf16* Wt, int K, int N, int Kp) {
    int total = N * Kp;
    qn_wconv<<<dim3((total + 255) / 256), dim3(256), 0, stream>>>(W, Wt, K, N, Kp);
  };
  wconv(se_w,  se_wT,  DD, HH, DD);
  wconv(key_w, key_wT, HH, HH, HH);
  wconv(q_w,   q_wT,   HH, HH, HH);
  wconv(sae_w, sae_wT, DD + AA, HH, KSAE);
  wconv(av_w,  av_wT,  HH, HH, HH);
  wconv(ca_w,  ca_wT,  DD, CC, DD);
  wconv(f1_w,  f1_wT,  CC + HH, CC, CC + HH);
  wconv(f2_w,  f2_wT,  CC, AA, CC);

  // 3) token-level GEMMs (bf16 WMMA, f32 accumulate, LDS-staged B panel)
  auto gemm = [&](const __bf16* A, const __bf16* Bt, const float* bias, void* C,
                  int M, int N, int K, int act, int obf) {
    qn_gemm<<<dim3(N / GT_N, M / GT_M), dim3(256), 0, stream>>>(A, Bt, bias, C,
                                                                M, N, K, act, obf);
  };
  gemm(states_bf, se_wT,  se_b,  s_embed, TT, HH, DD,     1, 1); // s_embed
  gemm(obsact_bf, sae_wT, sae_b, t1,      TT, HH, KSAE,   1, 1); // tanh(sae)
  gemm(s_embed,   key_wT, key_b, keyv,    TT, HH, HH,     0, 1); // key_obs
  gemm(s_embed,   q_wT,   q_b,   queryv,  TT, HH, HH,     0, 1); // query_obs
  gemm(t1,        av_wT,  av_b,  gbuf,    TT, HH, HH,     1, 0); // attn_vals (per node)
  gemm(states_bf, ca_wT,  ca_b,  currb,   TT, CC, DD,     1, 0); // curr

  // 4) per-batch attention + head
  float* out_value = (float*)d_out;
  float* out_qval  = out_value + (size_t)BB * NN * NN;
  float* out_wfull = out_qval  + (size_t)BB * NN;
  qn_attn<<<dim3(BB), dim3(256), 0, stream>>>(queryv, keyv, gbuf, currb,
                                              f1_wT, f1_b, f2_wT, f2_b,
                                              actions, policies,
                                              out_value, out_qval, out_wfull);
}
